// BiLSTM_CRF_48241072669050
// MI455X (gfx1250) — matrix operand order repaired and checked by reading, not compiled, not run on hardware
//
#include <hip/hip_runtime.h>
#include <hip/hip_bf16.h>

#define SEQ   8192
#define EMB   256
#define HID   256
#define HH    128       // per-direction hidden
#define G4    512       // 4*HH (PyTorch gate stack i,f,g,o)
#define NTAG  14
#define START_TAG 12
#define STOP_TAG  13
#define NEGV  (-10000.0f)
#define VCHUNK 128      // viterbi feats staging chunk (steps)

typedef __bf16 bf16_t;
typedef __attribute__((ext_vector_type(16))) __bf16 v16bf;
typedef __attribute__((ext_vector_type(8)))  __bf16 v8bf;
typedef __attribute__((ext_vector_type(8)))  float  v8f;

__device__ __forceinline__ float sigmoidf_(float x) { return 1.0f / (1.0f + __expf(-x)); }

// ---------------------------------------------------------------------------
// Kernel 1: convert Wih to bf16 (row-major [512,256]) and fuse biases.
// ---------------------------------------------------------------------------
__global__ __launch_bounds__(256) void cvt_kernel(
    const float* __restrict__ wih_f, const float* __restrict__ wih_b,
    const float* __restrict__ bih_f, const float* __restrict__ bhh_f,
    const float* __restrict__ bih_b, const float* __restrict__ bhh_b,
    bf16_t* __restrict__ wf, bf16_t* __restrict__ wb,
    float* __restrict__ bsf, float* __restrict__ bsb) {
  int idx = blockIdx.x * blockDim.x + threadIdx.x;
  if (idx < G4 * EMB) {
    wf[idx] = (bf16_t)wih_f[idx];
    wb[idx] = (bf16_t)wih_b[idx];
  }
  if (idx < G4) {
    bsf[idx] = bih_f[idx] + bhh_f[idx];
    bsb[idx] = bih_b[idx] + bhh_b[idx];
  }
}

// ---------------------------------------------------------------------------
// Kernel 2: gate pre-activations  G[t][g] = x[t] @ Wih^T + (bih+bhh)
// One block per 16-token tile; bf16 WMMA over K=256 (8 x 16x16x32 steps).
// ---------------------------------------------------------------------------
__global__ __launch_bounds__(256) void gates_kernel(
    const int*   __restrict__ sentence, const float* __restrict__ emb,
    const bf16_t* __restrict__ wf, const bf16_t* __restrict__ wb,
    const float* __restrict__ bsf, const float* __restrict__ bsb,
    float* __restrict__ Gf, float* __restrict__ Gb) {
  __shared__ bf16_t Abf[16][EMB];   // 8 KB staged bf16 activations
  const int tile = blockIdx.x;
  const int tid  = threadIdx.x;

  // Stage: gather 16 embedding rows, convert f32 -> bf16 into LDS.
  {
    int row = tid >> 4;
    int cb  = (tid & 15) * 16;
    int tok = sentence[tile * 16 + row];
    const float* src = emb + (size_t)tok * EMB + cb;
#pragma unroll
    for (int i = 0; i < 16; ++i) Abf[row][cb + i] = (bf16_t)src[i];
  }
  __syncthreads();

  const int lane  = tid & 31;
  const int wave  = tid >> 5;
  const int khalf = lane >> 4;   // which K-half this lane supplies
  const int nrow  = lane & 15;   // A row (M) / B column (N) within tile

  // 64 output N-tiles (32 fwd + 32 bwd) split across 8 waves.
#pragma unroll 1
  for (int nt = wave; nt < 64; nt += 8) {
    const int dir  = nt >> 5;
    const int ncol = (nt & 31) * 16 + nrow;       // gate index 0..511
    const bf16_t* __restrict__ W  = dir ? wb : wf;
    const float*  __restrict__ BS = dir ? bsb : bsf;
    float*        __restrict__ Go = dir ? Gb : Gf;
    const bf16_t* wrow = W + (size_t)ncol * EMB;
    __builtin_prefetch(wrow, 0, 1);               // global_prefetch_b8

    v8f acc = {};
#pragma unroll
    for (int kt = 0; kt < 8; ++kt) {
      // A fragment, contiguous K order (matches B):
      // lanes 0-15: K {0..15}; lanes 16-31: K {16..31}
      v8bf a0 = *(const v8bf*)&Abf[nrow][kt * 32 + khalf * 16];
      v8bf a1 = *(const v8bf*)&Abf[nrow][kt * 32 + khalf * 16 + 8];
      v16bf afrag;
#pragma unroll
      for (int i = 0; i < 8; ++i) { afrag[i] = a0[i]; afrag[i + 8] = a1[i]; }
      // B fragment: B[k][n] = Wih[n][k] -> contiguous 16 bf16 from weight row.
      v16bf bfrag = *(const v16bf*)(wrow + kt * 32 + khalf * 16);
      acc = __builtin_amdgcn_wmma_f32_16x16x32_bf16(
          false, afrag, false, bfrag, (short)0, acc, false, false);
    }
    // C layout: VGPR r -> M = r + 8*(lane>=16), N = lane&15.
    const float bias = BS[ncol];
    const int m0 = khalf * 8;
#pragma unroll
    for (int r = 0; r < 8; ++r)
      Go[(size_t)(tile * 16 + m0 + r) * G4 + ncol] = acc[r] + bias;
  }
}

// ---------------------------------------------------------------------------
// Kernel 3: the serial recurrences. 2 blocks (fwd / bwd), 512 threads each;
// thread g owns gate g: Whh row pinned in 32 float4 VGPR groups, h broadcast
// from LDS via b128 reads. One LDS round-trip + 2 barriers per step.
// ---------------------------------------------------------------------------
__global__ __launch_bounds__(512) void lstm_scan_kernel(
    const float* __restrict__ whh_f, const float* __restrict__ whh_b,
    const float* __restrict__ Gf, const float* __restrict__ Gb,
    float* __restrict__ hs_f, float* __restrict__ hs_b) {
  __shared__ __align__(16) float hsh[HH];
  __shared__ float gsh[G4];
  const int dir = blockIdx.x;           // 0 = forward, 1 = backward
  const int g   = threadIdx.x;          // gate index 0..511
  const float* __restrict__ whh = dir ? whh_b : whh_f;
  const float* __restrict__ G   = dir ? Gb : Gf;
  float*       __restrict__ hs  = dir ? hs_b : hs_f;

  float4 w4[HH / 4];
#pragma unroll
  for (int k = 0; k < HH / 4; ++k)
    w4[k] = *(const float4*)&whh[(size_t)g * HH + k * 4];

  float c = 0.0f;
  if (g < HH) hsh[g] = 0.0f;
  __syncthreads();

  for (int s = 0; s < SEQ; ++s) {
    const int t = dir ? (SEQ - 1 - s) : s;   // backward runs on reversed x
    float acc = G[(size_t)t * G4 + g];
    if (s + 1 < SEQ) {                       // stream-prefetch next step's row
      const int tn = dir ? (SEQ - 2 - s) : (s + 1);
      __builtin_prefetch(&G[(size_t)tn * G4 + g], 0, 1);
    }
#pragma unroll
    for (int k = 0; k < HH / 4; ++k) {
      float4 hv = *(const float4*)&hsh[k * 4];   // broadcast ds_load_b128
      acc += w4[k].x * hv.x + w4[k].y * hv.y + w4[k].z * hv.z + w4[k].w * hv.w;
    }
    gsh[g] = acc;
    __syncthreads();
    if (g < HH) {
      float iv = sigmoidf_(gsh[g]);
      float fv = sigmoidf_(gsh[HH + g]);
      float gv = tanhf(gsh[2 * HH + g]);
      float ov = sigmoidf_(gsh[3 * HH + g]);
      c = fv * c + iv * gv;
      float h = ov * tanhf(c);
      hsh[g] = h;
      hs[(size_t)t * HH + g] = h;            // hb already un-reversed via t
    }
    __syncthreads();
  }
}

// ---------------------------------------------------------------------------
// Kernel 4: tag projection  feats[t][tag] = [hf|hb] @ w_tag^T + b_tag
// ---------------------------------------------------------------------------
__global__ __launch_bounds__(256) void feats_kernel(
    const float* __restrict__ hs_f, const float* __restrict__ hs_b,
    const float* __restrict__ w_tag, const float* __restrict__ b_tag,
    float* __restrict__ feats) {
  int idx = blockIdx.x * blockDim.x + threadIdx.x;  // exactly SEQ*NTAG
  int t = idx / NTAG, tag = idx % NTAG;
  const float* wt = w_tag + (size_t)tag * HID;
  const float* hf = hs_f + (size_t)t * HH;
  const float* hb = hs_b + (size_t)t * HH;
  float acc = b_tag[tag];
#pragma unroll 8
  for (int j = 0; j < HH; ++j) acc += hf[j] * wt[j] + hb[j] * wt[HH + j];
  feats[idx] = acc;
}

// ---------------------------------------------------------------------------
// Kernel 5: Viterbi. One wave32. feats staged to LDS in 128-step chunks
// (coalesced float4 loads + prefetch) so the serial chain never waits on
// HBM; backpointers kept entirely in LDS for the backtrack.
// out[0]=score, out[1+t]=path[t].
// ---------------------------------------------------------------------------
__global__ __launch_bounds__(32) void viterbi_kernel(
    const float* __restrict__ feats, const float* __restrict__ trans,
    float* __restrict__ out) {
  extern __shared__ unsigned char smem[];
  float* fvs = (float*)smem;                       // 16 floats
  float* fbuf = (float*)(smem + 64);               // VCHUNK*NTAG floats
  unsigned char* bptr = smem + 64 + VCHUNK * NTAG * 4;  // SEQ*NTAG bytes
  const int lane = threadIdx.x;
  const int row  = (lane < NTAG) ? lane : 0;

  float tr[NTAG];
#pragma unroll
  for (int p = 0; p < NTAG; ++p) tr[p] = trans[row * NTAG + p];
  if (lane < NTAG) fvs[lane] = (lane == START_TAG) ? 0.0f : NEGV;
  __syncthreads();

  for (int t0 = 0; t0 < SEQ; t0 += VCHUNK) {
    // Cooperative coalesced stage of this chunk's feats into LDS.
    const float4* src = (const float4*)(feats + (size_t)t0 * NTAG);
    float4* dst = (float4*)fbuf;
    for (int i = lane; i < (VCHUNK * NTAG) / 4; i += 32) dst[i] = src[i];
    if (t0 + VCHUNK < SEQ)
      __builtin_prefetch(feats + (size_t)(t0 + VCHUNK) * NTAG + lane * 16, 0, 1);
    __syncthreads();

    for (int tt = 0; tt < VCHUNK; ++tt) {
      const int t = t0 + tt;
      float fvp[NTAG];
#pragma unroll
      for (int p = 0; p < NTAG; ++p) fvp[p] = fvs[p];
      float best = -3.0e38f; int barg = 0;
#pragma unroll
      for (int p = 0; p < NTAG; ++p) {
        float v = fvp[p] + tr[p];
        if (v > best) { best = v; barg = p; }   // strict > == argmax-first
      }
      __syncthreads();
      if (lane < NTAG) {
        fvs[lane] = best + fbuf[tt * NTAG + lane];
        bptr[t * NTAG + lane] = (unsigned char)barg;
      }
      __syncthreads();
    }
  }

  if (lane == 0) {
    float bestv = -3.0e38f; int bt = 0;
    for (int p = 0; p < NTAG; ++p) {
      float v = fvs[p] + trans[STOP_TAG * NTAG + p];
      if (v > bestv) { bestv = v; bt = p; }
    }
    out[0] = bestv;
    int tag = bt;
    for (int t = SEQ - 1; t >= 0; --t) {      // reverse scan emits carry
      out[1 + t] = (float)tag;
      tag = bptr[t * NTAG + tag];
    }
  }
}

// ---------------------------------------------------------------------------
extern "C" void kernel_launch(void* const* d_in, const int* in_sizes, int n_in,
                              void* d_out, int out_size, void* d_ws, size_t ws_size,
                              hipStream_t stream) {
  const int*   sentence = (const int*)  d_in[0];
  const float* emb      = (const float*)d_in[1];
  const float* wih_f    = (const float*)d_in[2];
  const float* whh_f    = (const float*)d_in[3];
  const float* bih_f    = (const float*)d_in[4];
  const float* bhh_f    = (const float*)d_in[5];
  const float* wih_b    = (const float*)d_in[6];
  const float* whh_b    = (const float*)d_in[7];
  const float* bih_b    = (const float*)d_in[8];
  const float* bhh_b    = (const float*)d_in[9];
  const float* w_tag    = (const float*)d_in[10];
  const float* b_tag    = (const float*)d_in[11];
  const float* trans    = (const float*)d_in[12];
  float* out = (float*)d_out;

  char* ws = (char*)d_ws;
  size_t off = 0;
  bf16_t* wfbf = (bf16_t*)(ws + off); off += (size_t)G4 * EMB * sizeof(bf16_t);
  bf16_t* wbbf = (bf16_t*)(ws + off); off += (size_t)G4 * EMB * sizeof(bf16_t);
  float* bsf   = (float*)(ws + off);  off += (size_t)G4 * sizeof(float);
  float* bsb   = (float*)(ws + off);  off += (size_t)G4 * sizeof(float);
  float* Gf    = (float*)(ws + off);  off += (size_t)SEQ * G4 * sizeof(float);
  float* Gb    = (float*)(ws + off);  off += (size_t)SEQ * G4 * sizeof(float);
  float* hsf   = (float*)(ws + off);  off += (size_t)SEQ * HH * sizeof(float);
  float* hsb   = (float*)(ws + off);  off += (size_t)SEQ * HH * sizeof(float);
  float* feats = (float*)(ws + off);  off += (size_t)SEQ * NTAG * sizeof(float);
  (void)ws_size; (void)in_sizes; (void)n_in; (void)out_size;

  cvt_kernel<<<(G4 * EMB + 255) / 256, 256, 0, stream>>>(
      wih_f, wih_b, bih_f, bhh_f, bih_b, bhh_b, wfbf, wbbf, bsf, bsb);
  gates_kernel<<<SEQ / 16, 256, 0, stream>>>(
      sentence, emb, wfbf, wbbf, bsf, bsb, Gf, Gb);
  lstm_scan_kernel<<<2, 512, 0, stream>>>(whh_f, whh_b, Gf, Gb, hsf, hsb);
  feats_kernel<<<(SEQ * NTAG) / 256, 256, 0, stream>>>(hsf, hsb, w_tag, b_tag, feats);
  viterbi_kernel<<<1, 32, 64 + VCHUNK * NTAG * 4 + SEQ * NTAG, stream>>>(
      feats, trans, out);
}
